// Token_QK_Attention_84078279786745
// MI455X (gfx1250) — matrix operand order repaired and checked by reading, not compiled, not run on hardware
//
#include <hip/hip_runtime.h>
#include <hip/hip_bf16.h>

// MI455X (gfx1250) fused spiking token-QK attention.
// GEMMs run as v_wmma_f32_16x16x32_bf16 (bf16 A/B, f32 accum): 16384 FLOP/instr
// vs 2048 for the f32 16x16x4 WMMA -> keeps the 51.6 GFLOP of conv1x1 work
// balanced against the 23.3 TB/s HBM roofline (~25us of mandatory traffic).
// d_out is reused as staging for k-spikes (kernel 2 writes, kernel 4 consumes
// and overwrites the identical region after an intra-block barrier).

typedef __bf16 bf16;
typedef __attribute__((ext_vector_type(16))) __bf16 v16bf;
typedef __attribute__((ext_vector_type(8)))  __bf16 v8bf;
typedef __attribute__((ext_vector_type(8)))  float  v8f;

#define T_DIM 4
#define B_DIM 32
#define C_DIM 256
#define N_DIM 1024
#define HEADS 8
#define HD 32
#define LXPAD 264   // LDS row pitch in bf16 (padded to dodge bank conflicts)

static __device__ __forceinline__ bf16 f2bf(float f) {
  unsigned u = __float_as_uint(f);
  unsigned r = (u + 0x7FFFu + ((u >> 16) & 1u)) >> 16;
  unsigned short s = (unsigned short)r;
  return __builtin_bit_cast(bf16, s);
}

// A fragment: weight rows (out-channels) x K=32 in-channels, row-major source.
// ISA layout: lanes<16 hold K {k0..k0+7, k0+16..k0+23}; lanes>=16 are +8.
static __device__ __forceinline__ v16bf load_a_frag(const bf16* w, int row, int k0, int koff) {
  const bf16* p = w + (size_t)row * C_DIM + k0 + koff;
  v8bf lo = *(const v8bf*)(p);
  v8bf hi = *(const v8bf*)(p + 16);
  v16bf a;
#pragma unroll
  for (int i = 0; i < 8; ++i) { a[i] = lo[i]; a[i + 8] = hi[i]; }
  return a;
}

// B fragment from transposed LDS tile lx[n_local][c]: lane holds column n,
// K {k0..k0+7, k0+16..k0+23} (lanes<16); lanes>=16 are +8.
static __device__ __forceinline__ v16bf load_b_frag(const bf16 (*lx)[LXPAD], int nloc, int k0, int kh) {
  const bf16* p = &lx[nloc][k0 + kh];
  v8bf lo = *(const v8bf*)(p);
  v8bf hi = *(const v8bf*)(p + 16);
  v16bf b;
#pragma unroll
  for (int i = 0; i < 8; ++i) { b[i] = lo[i]; b[i + 8] = hi[i]; }
  return b;
}

static __device__ __forceinline__ v8f wmma_bf16(v16bf a, v16bf b, v8f c) {
  return __builtin_amdgcn_wmma_f32_16x16x32_bf16(false, a, false, b, (short)0, c, false, false);
}

// ---------------------------------------------------------------- kernel 1
__global__ void prep_kernel(
    const float* __restrict__ qw, const float* __restrict__ kw, const float* __restrict__ pw,
    const float* __restrict__ qg, const float* __restrict__ qbeta,
    const float* __restrict__ qrm, const float* __restrict__ qrv,
    const float* __restrict__ kg, const float* __restrict__ kbeta,
    const float* __restrict__ krm, const float* __restrict__ krv,
    const float* __restrict__ pb, const float* __restrict__ pg,
    const float* __restrict__ pbeta, const float* __restrict__ prm, const float* __restrict__ prv,
    bf16* __restrict__ wq, bf16* __restrict__ wk, bf16* __restrict__ wp,
    float* __restrict__ qinv, float* __restrict__ qbias,
    float* __restrict__ kinv, float* __restrict__ kbias,
    float* __restrict__ pinv, float* __restrict__ pbias) {
  int idx = blockIdx.x * blockDim.x + threadIdx.x;
  int stride = gridDim.x * blockDim.x;
  for (int i = idx; i < C_DIM * C_DIM; i += stride) {
    wq[i] = f2bf(qw[i]);
    wk[i] = f2bf(kw[i]);
    wp[i] = f2bf(pw[i]);
  }
  if (idx < C_DIM) {
    float iq = qg[idx] * rsqrtf(qrv[idx] + 1e-5f);
    qinv[idx] = iq; qbias[idx] = qbeta[idx] - qrm[idx] * iq;
    float ik = kg[idx] * rsqrtf(krv[idx] + 1e-5f);
    kinv[idx] = ik; kbias[idx] = kbeta[idx] - krm[idx] * ik;
    float ip = pg[idx] * rsqrtf(prv[idx] + 1e-5f);
    pinv[idx] = ip; pbias[idx] = pbeta[idx] - prm[idx] * ip + pb[idx] * ip;  // proj_b folded
  }
}

// ---------------------------------------------------------------- kernel 2
// Q/K GEMM + BN + LIF(th=1) with membrane state in registers across T,
// per-head q_gate reduction, k spikes staged into d_out.
__global__ void __launch_bounds__(256) qk_kernel(
    const float* __restrict__ x, const bf16* __restrict__ wq, const bf16* __restrict__ wk,
    const float* __restrict__ qinv, const float* __restrict__ qbias,
    const float* __restrict__ kinv, const float* __restrict__ kbias,
    float* __restrict__ gate, float* __restrict__ kout) {
  const int b = blockIdx.x;
  const int n0 = blockIdx.y * 32;
  const int tid = threadIdx.x;
  const int wave = tid >> 5;        // == head id (HD==32 channels per wave)
  const int lane = tid & 31;
  const int m0 = wave * 32;
  const int koff = (lane & 16) ? 8 : 0;   // A-frag K sub-offset
  const int kh   = (lane & 16) ? 8 : 0;   // B-frag K sub-offset

  __shared__ bf16 lx[32][LXPAD];

  v8f vq[2][2], vk[2][2];
  const v8f vzero = {0.f, 0.f, 0.f, 0.f, 0.f, 0.f, 0.f, 0.f};
#pragma unroll
  for (int mt = 0; mt < 2; ++mt)
#pragma unroll
    for (int nt = 0; nt < 2; ++nt) { vq[mt][nt] = vzero; vk[mt][nt] = vzero; }

  for (int t = 0; t < T_DIM; ++t) {
    // stage x[t,b, c=tid, n0..n0+31] into LDS transposed [n][c] as bf16
    const float4* xs = (const float4*)(x + (((size_t)t * B_DIM + b) * C_DIM + tid) * N_DIM + n0);
#pragma unroll
    for (int j = 0; j < 8; ++j) {
      float4 v4 = xs[j];
      lx[4 * j + 0][tid] = f2bf(v4.x);
      lx[4 * j + 1][tid] = f2bf(v4.y);
      lx[4 * j + 2][tid] = f2bf(v4.z);
      lx[4 * j + 3][tid] = f2bf(v4.w);
    }
    __syncthreads();

    v8f cq[2][2], ck[2][2];
#pragma unroll
    for (int mt = 0; mt < 2; ++mt)
#pragma unroll
      for (int nt = 0; nt < 2; ++nt) { cq[mt][nt] = vzero; ck[mt][nt] = vzero; }

    const int arow0 = m0 + (lane & 15);
    for (int k0 = 0; k0 < C_DIM; k0 += 32) {
      v16bf aq0 = load_a_frag(wq, arow0,      k0, koff);
      v16bf aq1 = load_a_frag(wq, arow0 + 16, k0, koff);
      v16bf ak0 = load_a_frag(wk, arow0,      k0, koff);
      v16bf ak1 = load_a_frag(wk, arow0 + 16, k0, koff);
      v16bf b0  = load_b_frag(lx, (lane & 15),      k0, kh);
      v16bf b1  = load_b_frag(lx, 16 + (lane & 15), k0, kh);
      cq[0][0] = wmma_bf16(aq0, b0, cq[0][0]);
      cq[0][1] = wmma_bf16(aq0, b1, cq[0][1]);
      cq[1][0] = wmma_bf16(aq1, b0, cq[1][0]);
      cq[1][1] = wmma_bf16(aq1, b1, cq[1][1]);
      ck[0][0] = wmma_bf16(ak0, b0, ck[0][0]);
      ck[0][1] = wmma_bf16(ak0, b1, ck[0][1]);
      ck[1][0] = wmma_bf16(ak1, b0, ck[1][0]);
      ck[1][1] = wmma_bf16(ak1, b1, ck[1][1]);
    }

    // BN + LIF + per-head gate partial sums, emit k spikes
    float gsum0 = 0.f, gsum1 = 0.f;
#pragma unroll
    for (int mt = 0; mt < 2; ++mt) {
      const int chb = m0 + 16 * mt + ((lane & 16) ? 8 : 0);
#pragma unroll
      for (int r = 0; r < 8; ++r) {
        const int ch = chb + r;
        const float qi = qinv[ch], qb = qbias[ch];
        const float ki = kinv[ch], kb = kbias[ch];
#pragma unroll
        for (int nt = 0; nt < 2; ++nt) {
          float preq = cq[mt][nt][r] * qi + qb;
          float hq = 0.5f * (vq[mt][nt][r] + preq);          // tau = 2
          float sq = (hq >= 1.0f) ? 1.0f : 0.0f;             // v_th = 1
          vq[mt][nt][r] = hq * (1.0f - sq);
          if (nt == 0) gsum0 += sq; else gsum1 += sq;

          float prek = ck[mt][nt][r] * ki + kb;
          float hk = 0.5f * (vk[mt][nt][r] + prek);
          float sk = (hk >= 1.0f) ? 1.0f : 0.0f;
          vk[mt][nt][r] = hk * (1.0f - sk);
          const int n = n0 + nt * 16 + (lane & 15);
          kout[(((size_t)t * B_DIM + b) * C_DIM + ch) * N_DIM + n] = sk;
        }
      }
    }
    // combine lane pairs (L, L+16): together they cover all 32 head channels
    float g0 = gsum0 + __shfl_xor(gsum0, 16, 32);
    float g1 = gsum1 + __shfl_xor(gsum1, 16, 32);
    if ((lane & 16) == 0) {
      size_t base = (((size_t)t * B_DIM + b) * HEADS + wave) * N_DIM;
      gate[base + n0 + lane]      = g0 * (1.0f / 32.0f);
      gate[base + n0 + 16 + lane] = g1 * (1.0f / 32.0f);
    }
    __syncthreads();
  }
}

// ---------------------------------------------------------------- kernel 3
// Global fr reduction per head -> scale[8].
__global__ void fr_kernel(const float* __restrict__ gate, const float* __restrict__ frq,
                          float* __restrict__ scale) {
  const int head = blockIdx.x;
  const int tid = threadIdx.x;
  __shared__ float red[256];
  float s = 0.f;
  for (int i = tid; i < T_DIM * B_DIM * N_DIM; i += 256) {
    int tb = i >> 10;
    int n = i & (N_DIM - 1);
    s += gate[((size_t)tb * HEADS + head) * N_DIM + n];
  }
  red[tid] = s;
  __syncthreads();
  for (int o = 128; o > 0; o >>= 1) {
    if (tid < o) red[tid] += red[tid + o];
    __syncthreads();
  }
  if (tid == 0) {
    float mean = red[0] * (1.0f / (float)(T_DIM * B_DIM * N_DIM));
    float fr = frq[head] * 0.999f + mean * (1.0f - 0.999f);
    scale[head] = rsqrtf(fr * (float)HD + 1e-6f);
  }
}

// ---------------------------------------------------------------- kernel 4
// attn LIF(th=0.5) scan, xo = attn*k, proj GEMM + folded BN/bias + LIF(th=1).
// Reads k spikes from `out`, overwrites the same region with final spikes.
__global__ void __launch_bounds__(256) proj_kernel(
    const float* __restrict__ gate, const float* __restrict__ scalev,
    const bf16* __restrict__ wp,
    const float* __restrict__ pinv, const float* __restrict__ pbias,
    float* __restrict__ out) {
  const int b = blockIdx.x;
  const int n0 = blockIdx.y * 32;
  const int tid = threadIdx.x;
  const int lane = tid & 31;
  const int wave = tid >> 5;
  const int m0 = wave * 32;
  const int koff = (lane & 16) ? 8 : 0;
  const int kh   = (lane & 16) ? 8 : 0;

  __shared__ bf16 lx[32][LXPAD];
  __shared__ float lattn[T_DIM][HEADS][32];

  // attn LIF scan: one thread per (head, n_local)
  {
    const int head = tid >> 5;
    const int nl = tid & 31;
    const float sc = scalev[head];
    float v = 0.f;
#pragma unroll
    for (int t = 0; t < T_DIM; ++t) {
      float g = gate[(((size_t)t * B_DIM + b) * HEADS + head) * N_DIM + n0 + nl] * sc;
      float h = 0.5f * (v + g);
      float s = (h >= 0.5f) ? 1.0f : 0.0f;   // v_th = 0.5
      v = h * (1.0f - s);
      lattn[t][head][nl] = s;
    }
  }
  __syncthreads();

  v8f vp[2][2];
  const v8f vzero = {0.f, 0.f, 0.f, 0.f, 0.f, 0.f, 0.f, 0.f};
#pragma unroll
  for (int mt = 0; mt < 2; ++mt)
#pragma unroll
    for (int nt = 0; nt < 2; ++nt) vp[mt][nt] = vzero;

  for (int t = 0; t < T_DIM; ++t) {
    // stage xo = attn * k  (k spikes currently live in `out`)
    const int chead = tid >> 5;
    const float4* ks = (const float4*)(out + (((size_t)t * B_DIM + b) * C_DIM + tid) * N_DIM + n0);
#pragma unroll
    for (int j = 0; j < 8; ++j) {
      float4 v4 = ks[j];
      lx[4 * j + 0][tid] = f2bf(v4.x * lattn[t][chead][4 * j + 0]);
      lx[4 * j + 1][tid] = f2bf(v4.y * lattn[t][chead][4 * j + 1]);
      lx[4 * j + 2][tid] = f2bf(v4.z * lattn[t][chead][4 * j + 2]);
      lx[4 * j + 3][tid] = f2bf(v4.w * lattn[t][chead][4 * j + 3]);
    }
    __syncthreads();   // all reads of this t-slice done before any write below

    v8f cp[2][2];
#pragma unroll
    for (int mt = 0; mt < 2; ++mt)
#pragma unroll
      for (int nt = 0; nt < 2; ++nt) cp[mt][nt] = vzero;

    const int arow0 = m0 + (lane & 15);
    for (int k0 = 0; k0 < C_DIM; k0 += 32) {
      v16bf a0 = load_a_frag(wp, arow0,      k0, koff);
      v16bf a1 = load_a_frag(wp, arow0 + 16, k0, koff);
      v16bf b0 = load_b_frag(lx, (lane & 15),      k0, kh);
      v16bf b1 = load_b_frag(lx, 16 + (lane & 15), k0, kh);
      cp[0][0] = wmma_bf16(a0, b0, cp[0][0]);
      cp[0][1] = wmma_bf16(a0, b1, cp[0][1]);
      cp[1][0] = wmma_bf16(a1, b0, cp[1][0]);
      cp[1][1] = wmma_bf16(a1, b1, cp[1][1]);
    }

#pragma unroll
    for (int mt = 0; mt < 2; ++mt) {
      const int chb = m0 + 16 * mt + ((lane & 16) ? 8 : 0);
#pragma unroll
      for (int r = 0; r < 8; ++r) {
        const int ch = chb + r;
        const float pi = pinv[ch], pbv = pbias[ch];
#pragma unroll
        for (int nt = 0; nt < 2; ++nt) {
          float pre = cp[mt][nt][r] * pi + pbv;
          float h = 0.5f * (vp[mt][nt][r] + pre);
          float s = (h >= 1.0f) ? 1.0f : 0.0f;
          vp[mt][nt][r] = h * (1.0f - s);
          const int n = n0 + nt * 16 + (lane & 15);
          out[(((size_t)t * B_DIM + b) * C_DIM + ch) * N_DIM + n] = s;
        }
      }
    }
    __syncthreads();   // protect lx against next t's refill
  }
}

// ---------------------------------------------------------------- launch
extern "C" void kernel_launch(void* const* d_in, const int* in_sizes, int n_in,
                              void* d_out, int out_size, void* d_ws, size_t ws_size,
                              hipStream_t stream) {
  const float* x     = (const float*)d_in[0];
  const float* qw    = (const float*)d_in[1];
  const float* qg    = (const float*)d_in[2];
  const float* qbeta = (const float*)d_in[3];
  const float* qrm   = (const float*)d_in[4];
  const float* qrv   = (const float*)d_in[5];
  const float* kw    = (const float*)d_in[6];
  const float* kg    = (const float*)d_in[7];
  const float* kbeta = (const float*)d_in[8];
  const float* krm   = (const float*)d_in[9];
  const float* krv   = (const float*)d_in[10];
  const float* pw    = (const float*)d_in[11];
  const float* pb    = (const float*)d_in[12];
  const float* pg    = (const float*)d_in[13];
  const float* pbeta = (const float*)d_in[14];
  const float* prm   = (const float*)d_in[15];
  const float* prv   = (const float*)d_in[16];
  const float* frq   = (const float*)d_in[17];
  float* out = (float*)d_out;

  // ws layout: 3x bf16 weights (384 KB) | 6x256 BN params + 8 scales | gate (4 MB)
  bf16* wqb = (bf16*)d_ws;
  bf16* wkb = wqb + C_DIM * C_DIM;
  bf16* wpb = wkb + C_DIM * C_DIM;
  float* prms  = (float*)(wpb + C_DIM * C_DIM);
  float* qinv  = prms;
  float* qbias = prms + 256;
  float* kinv  = prms + 512;
  float* kbias = prms + 768;
  float* pinv  = prms + 1024;
  float* pbias = prms + 1280;
  float* scale = prms + 1536;
  float* gate  = prms + 2048;  // T*B*HEADS*N = 1M floats

  prep_kernel<<<96, 256, 0, stream>>>(qw, kw, pw,
                                      qg, qbeta, qrm, qrv,
                                      kg, kbeta, krm, krv,
                                      pb, pg, pbeta, prm, prv,
                                      wqb, wkb, wpb,
                                      qinv, qbias, kinv, kbias, pinv, pbias);

  qk_kernel<<<dim3(B_DIM, N_DIM / 32), 256, 0, stream>>>(
      x, wqb, wkb, qinv, qbias, kinv, kbias, gate, out);

  fr_kernel<<<HEADS, 256, 0, stream>>>(gate, frq, scale);

  proj_kernel<<<dim3(B_DIM, N_DIM / 32), 256, 0, stream>>>(
      gate, scale, wpb, pinv, pbias, out);
}
